// SpatialAttention_49615462204075
// MI455X (gfx1250) — matrix operand order repaired and checked by reading, not compiled, not run on hardware
//
#include <hip/hip_runtime.h>

// ---------------------------------------------------------------------------
// SpatialAttention for MI455X (gfx1250, wave32, WMMA 16x16x32 f16)
// B=2, T=2048, C=1024, H=16, Dh=64
// d_out layout (f32): out[B*T*C] | dist_energy | flop_energy | attn[B*H*T*T] | dists[T*T]
//
// v3: 16x64 wave GEMM tiles (no spills) with K-loop unrolled by 2 so 12 loads
//     cover 8 WMMAs per iteration (partial s_wait_loadcnt overlap, no
//     pipeline-copy registers). V kept transposed ([B,H,Dh,T]) so the PV
//     B-fragment is one contiguous 32B load per lane.
// ---------------------------------------------------------------------------

typedef _Float16 v16h __attribute__((ext_vector_type(16)));
typedef _Float16 v8h  __attribute__((ext_vector_type(8)));
typedef float    v8f  __attribute__((ext_vector_type(8)));

#define WMMA16(a, b, c) \
  __builtin_amdgcn_wmma_f32_16x16x32_f16(false, (a), false, (b), (short)0, (c), false, false)

#define BB 2
#define TT 2048
#define CC 1024
#define HH 16
#define DH 64
#define BH (BB * HH)          // 32
#define SROW 2052             // padded LDS row stride (floats)

// A-fragment (16x32 f16): lane holds row M=lane&15, K = (lane>>4)*16+e / +8
__device__ __forceinline__ v16h load_a_frag(const _Float16* __restrict__ row,
                                            int kc, int lh) {
  v8h lo = *(const v8h*)(row + kc + lh * 16);
  v8h hi = *(const v8h*)(row + kc + lh * 16 + 8);
  v16h a;
#pragma unroll
  for (int e = 0; e < 8; e++) { a[e] = lo[e]; a[e + 8] = hi[e]; }
  return a;
}

// ---------------------------------------------------------------- f32 -> f16
__global__ void cvt_f32_f16(const float* __restrict__ s, _Float16* __restrict__ d, int n) {
  for (int i = blockIdx.x * blockDim.x + threadIdx.x; i < n; i += gridDim.x * blockDim.x)
    d[i] = (_Float16)s[i];
}

// ------------------------------------------------- dists fill + scalar zeroing
__global__ void dists_init(float* __restrict__ dists, float* __restrict__ ed,
                           float* __restrict__ ef) {
  const long long n = (long long)TT * TT;
  for (long long idx = (long long)blockIdx.x * blockDim.x + threadIdx.x; idx < n;
       idx += (long long)gridDim.x * blockDim.x) {
    int i = (int)(idx >> 11), j = (int)(idx & (TT - 1));
    dists[idx] = fabsf((float)(i - j));
  }
  if (blockIdx.x == 0 && threadIdx.x == 0) { *ed = 0.0f; *ef = 0.0f; }
}

// --------------------------------------------------------------- QKV GEMM
// y[tok, o] = sum_c x[tok, c] * w[o, c]; one wave -> 16 (tokens) x 64 (features)
// q,k stored [B,H,T,Dh]; v stored transposed [B,H,Dh,T]
__global__ void qkv_gemm(const _Float16* __restrict__ X, const _Float16* __restrict__ W,
                         _Float16* __restrict__ Q, _Float16* __restrict__ Kb,
                         _Float16* __restrict__ Vt) {
  const int K = CC, NT = (3 * CC) / 64;        // 48 n-tiles of 64
  int wid  = (blockIdx.x * blockDim.x + threadIdx.x) >> 5;
  int lane = threadIdx.x & 31;
  int l15 = lane & 15, lh = lane >> 4;
  int mt = wid / NT, nt = wid % NT;
  int m0 = mt * 16, n0 = nt * 64;

  const _Float16* xrow = X + (size_t)(m0 + l15) * K;
  const _Float16* wrow = W + (size_t)(n0 + l15) * K;

  v8f acc[4] = {};
  for (int kc = 0; kc < K; kc += 64) {
    v16h aE = load_a_frag(xrow, kc, lh);
    v16h aO = load_a_frag(xrow, kc + 32, lh);
    v16h bE[4], bO[4];
#pragma unroll
    for (int t = 0; t < 4; t++)
      bE[t] = *(const v16h*)(wrow + (size_t)t * 16 * K + kc + lh * 16);
#pragma unroll
    for (int t = 0; t < 4; t++)
      bO[t] = *(const v16h*)(wrow + (size_t)t * 16 * K + kc + 32 + lh * 16);
#pragma unroll
    for (int t = 0; t < 4; t++) acc[t] = WMMA16(aE, bE[t], acc[t]);
#pragma unroll
    for (int t = 0; t < 4; t++) acc[t] = WMMA16(aO, bO[t], acc[t]);
  }

#pragma unroll
  for (int t = 0; t < 4; t++) {
#pragma unroll
    for (int r = 0; r < 8; r++) {
      int tok = m0 + r + lh * 8;
      int o = n0 + t * 16 + l15;
      int part = o >> 10;                      // 0=q 1=k 2=v
      int f = o & (CC - 1);
      int h = f >> 6, d = f & 63;
      int bi = tok >> 11, tl = tok & (TT - 1);
      _Float16 val = (_Float16)acc[t][r];
      size_t bhh = (size_t)(bi * HH + h);
      if (part == 0)      Q [(bhh * TT + tl) * DH + d] = val;
      else if (part == 1) Kb[(bhh * TT + tl) * DH + d] = val;
      else                Vt[(bhh * DH + d) * TT + tl] = val;   // transposed
    }
  }
}

// --------------------------------------------------------------- attention
// one block (128 thr = 4 waves) per (bh, 16-row tile); full logits row in LDS
__global__ void attn_kernel(const _Float16* __restrict__ Q, const _Float16* __restrict__ Kb,
                            const _Float16* __restrict__ Vt, const float* __restrict__ dsc,
                            float* __restrict__ attn_out, float* __restrict__ e_dist,
                            float* __restrict__ e_flop, _Float16* __restrict__ outPre) {
  extern __shared__ char smem[];
  float* S      = (float*)smem;                // 16 * SROW
  float* sOut   = S + 16 * SROW;               // 16 * 68
  float* red    = sOut + 16 * 68;              // 128
  float* rowMax = red + 128;                   // 16
  float* rowInv = rowMax + 16;                 // 16

  int mt = blockIdx.x & 127;                   // T/16 tiles
  int bh = blockIdx.x >> 7;
  int t0 = mt * 16;
  int tid = threadIdx.x;
  int w = tid >> 5, lane = tid & 31;
  int l15 = lane & 15, lh = lane >> 4;
  float dscale = dsc[bh & (HH - 1)];
  const float scale = 0.125f;                  // 1/sqrt(64)
  int nj = mt + 1;                             // column tiles touching causal region
  int ncols = nj * 16;

  // q A-fragments for K=0..31 and K=32..63
  const _Float16* qrow = Q + ((size_t)bh * TT + t0 + l15) * DH;
  v16h a0 = load_a_frag(qrow, 0, lh);
  v16h a1 = load_a_frag(qrow, 32, lh);

  // phase 1: logits (biased, masked) into LDS; waves stripe column tiles
  for (int j = w; j < nj; j += 4) {
    int j0 = j * 16;
    const _Float16* krow = Kb + ((size_t)bh * TT + j0 + l15) * DH;
    v16h b0 = *(const v16h*)(krow + lh * 16);
    v16h b1 = *(const v16h*)(krow + 32 + lh * 16);
    v8f s = {};
    s = WMMA16(a0, b0, s);
    s = WMMA16(a1, b1, s);
#pragma unroll
    for (int r = 0; r < 8; r++) {
      int m = r + lh * 8;
      int i = t0 + m;
      int c = j0 + l15;
      float val = (c > i) ? -__builtin_inff()
                          : (s[r] * scale - dscale * fabsf((float)(i - c)));
      S[m * SROW + c] = val;
    }
  }
  // pad tile to even K for the PV pass; zero the PV accumulator
  if (nj & 1) {
    for (int idx = tid; idx < 256; idx += 128)
      S[(idx >> 4) * SROW + ncols + (idx & 15)] = 0.0f;
  }
  for (int idx = tid; idx < 16 * 68; idx += 128) sOut[idx] = 0.0f;
  __syncthreads();

  // phase 2: softmax (8 threads per row)
  int row = tid >> 3, slot = tid & 7;
  float mx = -__builtin_inff();
  for (int c = slot; c < ncols; c += 8) mx = fmaxf(mx, S[row * SROW + c]);
  red[row * 8 + slot] = mx;
  __syncthreads();
  if (slot == 0) {
    float m2 = red[row * 8];
#pragma unroll
    for (int k = 1; k < 8; k++) m2 = fmaxf(m2, red[row * 8 + k]);
    rowMax[row] = m2;
  }
  __syncthreads();
  float rm = rowMax[row];
  float sm = 0.0f;
  for (int c = slot; c < ncols; c += 8) {
    float p = __expf(S[row * SROW + c] - rm);
    S[row * SROW + c] = p;                     // unnormalized prob
    sm += p;
  }
  red[row * 8 + slot] = sm;
  __syncthreads();
  if (slot == 0) {
    float z = 0.0f;
#pragma unroll
    for (int k = 0; k < 8; k++) z += red[row * 8 + k];
    rowInv[row] = 1.0f / z;
  }
  __syncthreads();

  // phase 2b: coalesced attn write (zeros past causal range) + energy partials
  float ed = 0.0f, ef = 0.0f;
  float* attnBase = attn_out + (size_t)bh * TT * TT + (size_t)t0 * TT;
  for (int idx = tid; idx < 16 * TT; idx += 128) {
    int m = idx >> 11, c = idx & (TT - 1);
    float ap = (c < ncols) ? S[m * SROW + c] * rowInv[m] : 0.0f;
    attnBase[(size_t)m * TT + c] = ap;
    int i = t0 + m;
    ed += ap * fabsf((float)(i - c));
    ef -= ap * __logf(ap + 1e-9f);
  }
  __syncthreads();
  red[tid] = ed;
  __syncthreads();
  for (int s2 = 64; s2 > 0; s2 >>= 1) { if (tid < s2) red[tid] += red[tid + s2]; __syncthreads(); }
  if (tid == 0) atomicAdd(e_dist, red[0] * (1.0f / (float)(BB * HH * TT)));
  __syncthreads();
  red[tid] = ef;
  __syncthreads();
  for (int s2 = 64; s2 > 0; s2 >>= 1) { if (tid < s2) red[tid] += red[tid + s2]; __syncthreads(); }
  if (tid == 0) atomicAdd(e_flop, red[0] * (1.0f / (float)(BB * HH * TT)));
  __syncthreads();

  // phase 3: out = (P/Z) @ V via WMMA on unnormalized P, scale by 1/Z at end.
  // B-fragment reads Vt[bh, d, t]: contiguous along t -> single 32B load/lane.
  int nsteps = ((nj + 1) & ~1) * 16 / 32;      // ceil(nj/2)
  v8f acc[4] = {};
  for (int ks = w; ks < nsteps; ks += 4) {
    int c0 = ks * 32;
    v16h a;
#pragma unroll
    for (int e = 0; e < 8; e++) {
      a[e]     = (_Float16)S[l15 * SROW + c0 + lh * 16 + e];
      a[e + 8] = (_Float16)S[l15 * SROW + c0 + lh * 16 + 8 + e];
    }
#pragma unroll
    for (int t = 0; t < 4; t++) {
      v16h b = *(const v16h*)(Vt + ((size_t)bh * DH + t * 16 + l15) * TT + c0 + lh * 16);
      acc[t] = WMMA16(a, b, acc[t]);
    }
  }
#pragma unroll
  for (int t = 0; t < 4; t++)
#pragma unroll
    for (int r = 0; r < 8; r++)
      atomicAdd(&sOut[(r + lh * 8) * 68 + t * 16 + l15], acc[t][r]);
  __syncthreads();

  int b_ = bh >> 4, h_ = bh & (HH - 1);
  for (int idx = tid; idx < 16 * 64; idx += 128) {
    int m = idx >> 6, d = idx & 63;
    float val = sOut[m * 68 + d] * rowInv[m];
    outPre[((size_t)(b_ * TT + t0 + m)) * CC + h_ * DH + d] = (_Float16)val;
  }
}

// --------------------------------------------------------------- projection
// one wave -> 16 (tokens) x 64 (features), K-loop unrolled by 2
__global__ void proj_gemm(const _Float16* __restrict__ X, const _Float16* __restrict__ W,
                          float* __restrict__ Y) {
  const int K = CC, NT = CC / 64;              // 16 n-tiles
  int wid  = (blockIdx.x * blockDim.x + threadIdx.x) >> 5;
  int lane = threadIdx.x & 31;
  int l15 = lane & 15, lh = lane >> 4;
  int mt = wid / NT, nt = wid % NT;
  int m0 = mt * 16, n0 = nt * 64;

  const _Float16* xrow = X + (size_t)(m0 + l15) * K;
  const _Float16* wrow = W + (size_t)(n0 + l15) * K;

  v8f acc[4] = {};
  for (int kc = 0; kc < K; kc += 64) {
    v16h aE = load_a_frag(xrow, kc, lh);
    v16h aO = load_a_frag(xrow, kc + 32, lh);
    v16h bE[4], bO[4];
#pragma unroll
    for (int t = 0; t < 4; t++)
      bE[t] = *(const v16h*)(wrow + (size_t)t * 16 * K + kc + lh * 16);
#pragma unroll
    for (int t = 0; t < 4; t++)
      bO[t] = *(const v16h*)(wrow + (size_t)t * 16 * K + kc + 32 + lh * 16);
#pragma unroll
    for (int t = 0; t < 4; t++) acc[t] = WMMA16(aE, bE[t], acc[t]);
#pragma unroll
    for (int t = 0; t < 4; t++) acc[t] = WMMA16(aO, bO[t], acc[t]);
  }

#pragma unroll
  for (int t = 0; t < 4; t++)
#pragma unroll
    for (int r = 0; r < 8; r++) {
      int tok = m0 + r + lh * 8;
      int o = n0 + t * 16 + l15;
      Y[(size_t)tok * CC + o] = acc[t][r];
    }
}

// ---------------------------------------------------------------------------
extern "C" void kernel_launch(void* const* d_in, const int* in_sizes, int n_in,
                              void* d_out, int out_size, void* d_ws, size_t ws_size,
                              hipStream_t stream) {
  const float* x      = (const float*)d_in[0];   // [B,T,C]
  const float* qkv_w  = (const float*)d_in[1];   // [3C,C]
  const float* proj_w = (const float*)d_in[2];   // [C,C]
  const float* dsc    = (const float*)d_in[3];   // [H]

  float* out = (float*)d_out;
  float* out_main = out;                                       // B*T*C
  float* e_dist   = out + (size_t)BB * TT * CC;                // scalar
  float* e_flop   = e_dist + 1;                                // scalar
  float* attn     = e_flop + 1;                                // B*H*T*T
  float* dists    = attn + (size_t)BB * HH * TT * TT;          // T*T

  char* ws = (char*)d_ws;
  _Float16* xh    = (_Float16*)(ws);                    // B*T*C      (8 MB)
  _Float16* wqkv  = (_Float16*)(ws + 8388608);          // 3C*C       (6 MB)
  _Float16* wproj = (_Float16*)(ws + 14680064);         // C*C        (2 MB)
  _Float16* qh    = (_Float16*)(ws + 16777216);         // [B,H,T,Dh] (8 MB)
  _Float16* kh    = (_Float16*)(ws + 25165824);         // [B,H,T,Dh]
  _Float16* vth   = (_Float16*)(ws + 33554432);         // [B,H,Dh,T] (transposed)
  _Float16* op    = (_Float16*)(ws + 41943040);         // [B,T,C] pre-proj

  cvt_f32_f16<<<1024, 256, 0, stream>>>(x, xh, BB * TT * CC);
  cvt_f32_f16<<<1024, 256, 0, stream>>>(qkv_w, wqkv, 3 * CC * CC);
  cvt_f32_f16<<<512, 256, 0, stream>>>(proj_w, wproj, CC * CC);
  dists_init<<<1024, 256, 0, stream>>>(dists, e_dist, e_flop);

  // 256 m-tiles * 48 n-tiles = 12288 waves / 4 waves per block
  qkv_gemm<<<3072, 128, 0, stream>>>(xh, wqkv, qh, kh, vth);

  size_t lds = (size_t)(16 * SROW + 16 * 68 + 128 + 32) * sizeof(float);  // ~136 KB
  attn_kernel<<<BH * (TT / 16), 128, lds, stream>>>(qh, kh, vth, dsc, attn,
                                                    e_dist, e_flop, op);

  // 256 m-tiles * 16 n-tiles = 4096 waves / 4 per block
  proj_gemm<<<1024, 128, 0, stream>>>(op, wproj, out_main);
}
